// extVQ_9586367004962
// MI455X (gfx1250) — matrix operand order repaired and checked by reading, not compiled, not run on hardware
//
#include <hip/hip_runtime.h>
#include <hip/hip_bf16.h>
#include <math.h>

// ---------------------------------------------------------------------------
// VQ-VAE vector quantization for MI455X (gfx1250, wave32, WMMA).
//   N = 131072 rows (B*H*W), D = 64, codes = up to 1024 (2*K).
// Distance argmin via augmented-K WMMA: score = x.c - ||c||^2/2 comes straight
// out of v_wmma_f32_16x16x32_bf16 (K = 96: 64 data + 2 norm-correction + pad).
// ---------------------------------------------------------------------------

typedef __bf16 v16bf __attribute__((ext_vector_type(16)));
typedef __bf16 v8bf  __attribute__((ext_vector_type(8)));
typedef float  v8f   __attribute__((ext_vector_type(8)));

#define NROWS      131072      // B*H*W
#define HW         4096        // H*W
#define CHW        262144      // C*H*W
#define DDIM       64
#define KPAD       96          // augmented, padded K (3 x 32)
#define KMAX       1024
#define CHUNK_ROWS 128         // codes per LDS chunk
#define ROWS_PER_BLOCK 128     // 8 waves * 16 rows

// ---- A fragment: 16-bit A 16x32 layout (ISA 7.12.2) -----------------------
// lanes 0-15 : VGPR0-3 = K 0..7,  VGPR4-7 = K 16..23
// lanes 16-31: VGPR0-3 = K 8..15, VGPR4-7 = K 24..31
__device__ inline v16bf load_frag_a(const __bf16* p /* &row[kc*32] */, int half) {
  v8bf lo = *(const v8bf*)(p + 8 * half);
  v8bf hi = *(const v8bf*)(p + 16 + 8 * half);
  return __builtin_shufflevector(lo, hi, 0,1,2,3,4,5,6,7,8,9,10,11,12,13,14,15);
}

// ---- B fragment: 16-bit B 32x16 layout (same K order as A) ----------------
// lanes 0-15 : K 0..7, 16..23   lanes 16-31 : K 8..15, 24..31   (N = lane & 15)
__device__ inline v16bf load_frag_b(const __bf16* p /* &coderow[kc*32] */, int half) {
  v8bf lo = *(const v8bf*)(p + 8 * half);
  v8bf hi = *(const v8bf*)(p + 16 + 8 * half);
  return __builtin_shufflevector(lo, hi, 0,1,2,3,4,5,6,7,8,9,10,11,12,13,14,15);
}

// ===========================================================================
// Kernel 0: codebook -> bf16 rows padded to KPAD with the norm correction:
//   row[0..63]  = bf16(code)
//   row[64]     = bf16(-||c||^2/2)            (hi part)
//   row[65]     = bf16(residual of the above)  (lo part)
//   row[66..95] = 0
// Also zeroes counts / SSE.
// ===========================================================================
__global__ __launch_bounds__(256) void vq_prep_kernel(
    const float* __restrict__ cb0, const float* __restrict__ cb1,
    __bf16* __restrict__ codesB, int* __restrict__ counts,
    double* __restrict__ sse)
{
  int row = blockIdx.x * blockDim.x + threadIdx.x;  // 0..1023
  if (row < KMAX) {
    const float* src = (row < 512) ? (cb0 + row * DDIM) : (cb1 + (row - 512) * DDIM);
    __bf16* dst = codesB + (size_t)row * KPAD;
    float acc = 0.0f;
    #pragma unroll
    for (int d = 0; d < DDIM; ++d) {
      float f = src[d];
      acc += f * f;
      dst[d] = (__bf16)f;
    }
    float corr = -0.5f * acc;
    __bf16 hi  = (__bf16)corr;
    dst[DDIM]     = hi;
    dst[DDIM + 1] = (__bf16)(corr - (float)hi);
    #pragma unroll
    for (int d = DDIM + 2; d < KPAD; ++d) dst[d] = (__bf16)0.0f;
    counts[row] = 0;
  }
  if (row == 0) *sse = 0.0;
}

// ===========================================================================
// Kernel 1: score argmax via v_wmma_f32_16x16x32_bf16 (3 chained, K = 96)
// grid = 1024 blocks x 256 threads (8 waves); each wave owns 16 x-rows.
// score = x.c - ||c||^2/2 ; argmax(score) == argmin(distance)
// ===========================================================================
__global__ __launch_bounds__(256) void vq_argmin_kernel(
    const float* __restrict__ inputs, const __bf16* __restrict__ codesB,
    const int* __restrict__ idxSel, int* __restrict__ bestIdx)
{
  __shared__ __bf16 ldsX[8][16 * KPAD];         // 24 KB : per-wave x tiles
  __shared__ __bf16 ldsC[CHUNK_ROWS * KPAD];    // 24 KB : codebook chunk

  const int Kt   = (*idxSel == 0) ? 512 : 1024;
  const int tid  = threadIdx.x;
  const int wave = tid >> 5;
  const int lane = tid & 31;
  const int half = lane >> 4;
  const int r    = lane & 15;

  // ---- stage x tile: NCHW -> row-major [16][96] bf16 (augmented) ----
  const int rowBase = blockIdx.x * ROWS_PER_BLOCK + wave * 16;
  const int b  = rowBase >> 12;       // tile never crosses a batch boundary
  const int hw = rowBase & (HW - 1);
  const float* xp = inputs + (size_t)b * CHW + hw + r;   // lane's row
  #pragma unroll
  for (int i = 0; i < KPAD / 2; ++i) {
    int d = half * (KPAD / 2) + i;    // two lanes per row cover d = 0..95
    __bf16 v;
    if (d < DDIM)          v = (__bf16)xp[(size_t)d * HW];
    else if (d < DDIM + 2) v = (__bf16)1.0f;    // picks up the norm correction
    else                   v = (__bf16)0.0f;
    ldsX[wave][r * KPAD + d] = v;
  }
  __syncthreads();

  // ---- A fragments (K = 0..31, 32..63, 64..95) ----
  const __bf16* xrow = &ldsX[wave][r * KPAD];
  v16bf a0 = load_frag_a(xrow,      half);
  v16bf a1 = load_frag_a(xrow + 32, half);
  v16bf a2 = load_frag_a(xrow + 64, half);

  float bs[8];
  int   bi[8];
  #pragma unroll
  for (int v = 0; v < 8; ++v) { bs[v] = -3.4e38f; bi[v] = 0; }

  const unsigned ldsCBase = (unsigned)(size_t)(void*)&ldsC[0];
  const int nChunks = Kt / CHUNK_ROWS;
  for (int ch = 0; ch < nChunks; ++ch) {
    __syncthreads();                              // previous chunk fully consumed
    {   // async LDS fill: 128 rows x 96 bf16 = 24 KB, b128 per lane
      const unsigned chunkByteOff = (unsigned)ch * CHUNK_ROWS * KPAD * 2;
      for (int i = tid; i < (CHUNK_ROWS * KPAD) / 8; i += 256) {
        unsigned lo = ldsCBase + (unsigned)i * 16u;
        unsigned go = chunkByteOff + (unsigned)i * 16u;
        asm volatile("global_load_async_to_lds_b128 %0, %1, %2"
                     :: "v"(lo), "v"(go), "s"(codesB) : "memory");
      }
      asm volatile("s_wait_asynccnt 0x0" ::: "memory");
    }
    __syncthreads();

    const int cbase = ch * CHUNK_ROWS;
    #pragma unroll 2
    for (int tl = 0; tl < CHUNK_ROWS / 16; ++tl) {
      const __bf16* crow = &ldsC[(tl * 16 + r) * KPAD];   // code row N = lane&15
      v16bf fb0 = load_frag_b(crow,      half);
      v16bf fb1 = load_frag_b(crow + 32, half);
      v16bf fb2 = load_frag_b(crow + 64, half);

      v8f c = {0.f, 0.f, 0.f, 0.f, 0.f, 0.f, 0.f, 0.f};
      c = __builtin_amdgcn_wmma_f32_16x16x32_bf16(false, a0, false, fb0,
                                                  (short)0, c, false, false);
      c = __builtin_amdgcn_wmma_f32_16x16x32_bf16(false, a1, false, fb1,
                                                  (short)0, c, false, false);
      c = __builtin_amdgcn_wmma_f32_16x16x32_bf16(false, a2, false, fb2,
                                                  (short)0, c, false, false);

      const int ci = cbase + tl * 16 + r;         // this lane's code index (N)
      #pragma unroll
      for (int v = 0; v < 8; ++v) {               // row M = v + 8*half
        float s = c[v];
        if (s > bs[v]) { bs[v] = s; bi[v] = ci; }
      }
    }
  }

  // ---- reduce across the 16 lanes sharing each row (tie -> lowest index) ----
  #pragma unroll
  for (int v = 0; v < 8; ++v) {
    float s = bs[v];
    int   i = bi[v];
    #pragma unroll
    for (int m = 8; m >= 1; m >>= 1) {            // xor within lane&15 group
      float os = __shfl_xor(s, m, 32);
      int   oi = __shfl_xor(i, m, 32);
      if (os > s || (os == s && oi < i)) { s = os; i = oi; }
    }
    if (r == 0) bestIdx[rowBase + v + 8 * half] = i;
  }
}

// ===========================================================================
// Kernel 2: exact f32 gather + straight-through output (NCHW) + SSE + histogram
// ===========================================================================
__global__ __launch_bounds__(256) void vq_gather_kernel(
    const float* __restrict__ inputs, const float* __restrict__ cb0,
    const float* __restrict__ cb1, const int* __restrict__ bestIdx,
    float* __restrict__ out, int* __restrict__ counts, double* __restrict__ sse)
{
  const int n = blockIdx.x * 256 + threadIdx.x;
  const int k = bestIdx[n];
  const float* crow = (k < 512) ? (cb0 + (size_t)k * DDIM)
                                : (cb1 + (size_t)(k - 512) * DDIM);
  const int b  = n >> 12;
  const int hw = n & (HW - 1);
  const float* xp = inputs + (size_t)b * CHW + hw;
  float*       op = out    + (size_t)b * CHW + hw;

  float acc = 0.0f;
  #pragma unroll
  for (int d = 0; d < DDIM; ++d) {
    float x    = xp[(size_t)d * HW];
    float q    = crow[d];
    float diff = q - x;
    op[(size_t)d * HW] = x + diff;               // straight-through: x + sg(q - x)
    acc += diff * diff;
  }
  atomicAdd(&counts[k], 1);

  #pragma unroll
  for (int m = 16; m >= 1; m >>= 1) acc += __shfl_down(acc, m, 32);
  if ((threadIdx.x & 31) == 0) atomicAdd(sse, (double)acc);
}

// ===========================================================================
// Kernel 3: loss + perplexity
// ===========================================================================
__global__ __launch_bounds__(1024) void vq_final_kernel(
    const int* __restrict__ counts, const double* __restrict__ sse,
    const int* __restrict__ idxSel, float* __restrict__ outTail)
{
  __shared__ float red[32];
  const int Kt = (*idxSel == 0) ? 512 : 1024;
  const int t  = threadIdx.x;

  float term = 0.0f;
  if (t < Kt) {
    float p = (float)counts[t] * (1.0f / (float)NROWS);
    term = p * logf(p + 1e-10f);
  }
  #pragma unroll
  for (int m = 16; m >= 1; m >>= 1) term += __shfl_down(term, m, 32);
  if ((t & 31) == 0) red[t >> 5] = term;
  __syncthreads();
  if (t < 32) {
    float v = red[t];
    #pragma unroll
    for (int m = 16; m >= 1; m >>= 1) v += __shfl_down(v, m, 32);
    if (t == 0) {
      // q_latent_loss == e_latent_loss == mean((q-x)^2); loss = 1.25 * mse
      outTail[0] = (float)(1.25 * (*sse) / (double)(NROWS * DDIM));
      outTail[1] = expf(-v);
    }
  }
}

// ===========================================================================
// Launcher
// ===========================================================================
extern "C" void kernel_launch(void* const* d_in, const int* in_sizes, int n_in,
                              void* d_out, int out_size, void* d_ws, size_t ws_size,
                              hipStream_t stream) {
  (void)in_sizes; (void)n_in; (void)ws_size;
  const float* inputs = (const float*)d_in[0];
  const float* cb0    = (const float*)d_in[1];
  const float* cb1    = (const float*)d_in[2];
  const int*   idxSel = (const int*)d_in[3];
  float*       out    = (float*)d_out;

  // workspace layout
  char*   ws      = (char*)d_ws;
  __bf16* codesB  = (__bf16*)(ws);                        // 1024*96*2 = 196608 B
  int*    counts  = (int*)(ws + 196608);                  //   4096 B
  double* sse     = (double*)(ws + 200704);               //      8 B
  int*    bestIdx = (int*)(ws + 200712);                  // 524288 B

  vq_prep_kernel<<<4, 256, 0, stream>>>(cb0, cb1, codesB, counts, sse);
  vq_argmin_kernel<<<NROWS / ROWS_PER_BLOCK, 256, 0, stream>>>(
      inputs, codesB, idxSel, bestIdx);
  vq_gather_kernel<<<NROWS / 256, 256, 0, stream>>>(
      inputs, cb0, cb1, bestIdx, out, counts, sse);
  vq_final_kernel<<<1, 1024, 0, stream>>>(counts, sse, idxSel,
                                          out + (out_size - 2));
}
